// RPEEncoderLayer_28939489641154
// MI455X (gfx1250) — matrix operand order repaired and checked by reading, not compiled, not run on hardware
//
#include <hip/hip_runtime.h>
#include <hip/hip_bf16.h>

typedef _Float16 half8 __attribute__((ext_vector_type(8)));
typedef _Float16 v16h  __attribute__((ext_vector_type(16)));
typedef float    v8f   __attribute__((ext_vector_type(8)));

#define CAT16 0,1,2,3,4,5,6,7,8,9,10,11,12,13,14,15

static __device__ __forceinline__ v16h cat8(half8 lo, half8 hi) {
    return __builtin_shufflevector(lo, hi, CAT16);
}

static __device__ __forceinline__ float gelu_exact(float x) {
    return 0.5f * x * (1.0f + erff(x * 0.70710678118654752f));
}

// ---------------- prep kernels ----------------

__global__ void cvt_f32_f16(const float* __restrict__ src, _Float16* __restrict__ dst, int n) {
    int i = blockIdx.x * 256 + threadIdx.x;
    if (i < n) dst[i] = (_Float16)src[i];
}

__global__ void prep_params(const float* __restrict__ means, const float* __restrict__ stds,
                            float* __restrict__ mean_o, float* __restrict__ istd_o,
                            float* __restrict__ scl_o) {
    int i = blockIdx.x * 256 + threadIdx.x;   // 512 total (R*K)
    if (i < 512) {
        float sd   = fabsf(stds[i]) + 0.01f;
        float s2pi = sqrtf(2.0f * 3.14159f);  // matches reference constant
        mean_o[i] = means[i];
        istd_o[i] = 1.0f / sd;
        scl_o[i]  = 1.0f / (s2pi * sd);
    }
}

// ---------------- fused main kernel ----------------
// Grid: T/256 blocks of 256 threads (8 waves). Each wave owns 32 consecutive tokens.
// LDS: [0,256K)    g tile : 8 waves x 32 rows x 512 f16 (wave-private)
//      [256K,288K) h tile : 8 waves x 32 rows x 64  f16 (wave-private)
// 288KB LDS -> 1 block/WGP -> 2 waves/SIMD. GEMM1 uses a 2(M)x8(N) register
// tile over 128-col H chunks: A frags (LDS) reused 8x, B frags (global/WGP$)
// reused 2x. A __syncthreads() per chunk keeps all 8 waves on the same W1
// slab; the next slab is prefetched during the gelu/GEMM2 tail.

__global__ void __launch_bounds__(256)
__attribute__((amdgpu_waves_per_eu(2)))
rpe_fused(const float* __restrict__ rpe,
          const _Float16* __restrict__ W1h,   // [512][512] f16 row-major (h, f)
          const float* __restrict__ b1,
          const _Float16* __restrict__ W2h,   // [32][512] f16 row-major (o, h)
          const float* __restrict__ b2,
          const float* __restrict__ gmean,
          const float* __restrict__ gistd,
          const float* __restrict__ gscl,
          float* __restrict__ out) {
    __shared__ alignas(32) unsigned char smem[262144 + 32768];

    const int ln  = threadIdx.x & 31;
    const int wv  = threadIdx.x >> 5;
    const int col = ln & 15;        // WMMA column / row-in-block index
    const int khi = ln >> 4;        // lane-half selector

    _Float16* gbuf = (_Float16*)smem + wv * (32 * 512);
    _Float16* hbuf = (_Float16*)(smem + 262144) + wv * (32 * 64);
    float*    obuf = (float*)(void*)gbuf;     // aliases g region (dead by epilogue)

    const int tok0 = blockIdx.x * 256 + wv * 32;

    // ---- Phase 1: Gaussian features -> LDS (f16), wave-private ----
    {
        const int f0 = ln * 16;     // this lane's 16 contiguous features
        const int r  = f0 >> 7;     // which of the R=4 rpe channels
        float mean[16], istd[16], scl[16];
#pragma unroll
        for (int j = 0; j < 16; ++j) {
            mean[j] = gmean[f0 + j];
            istd[j] = gistd[f0 + j];
            scl[j]  = gscl[f0 + j];
        }
#pragma unroll 1
        for (int m = 0; m < 32; ++m) {
            float x = rpe[(size_t)(tok0 + m) * 4 + r];
            v16h val;
#pragma unroll
            for (int j = 0; j < 16; ++j) {
                float u = (x - mean[j]) * istd[j];
                val[j] = (_Float16)(scl[j] * __expf(-0.5f * u * u));
            }
            *(v16h*)(gbuf + m * 512 + f0) = val;   // 32B store -> 2x ds_store_b128
        }
    }

    // ---- Phase 2: fused GEMM1(gelu) -> GEMM2, accumulate over H ----
    v8f acc2[4] = {v8f{}, v8f{}, v8f{}, v8f{}};   // [mb*2+hc]: 32 rows x 32 heads

#pragma unroll 1
    for (int nc8 = 0; nc8 < 4; ++nc8) {           // H in chunks of 128
        __syncthreads();                          // pace waves onto same W1 slab

        // --- GEMM1: 32 rows x 128 cols, register tile 2(M) x 8(N) ---
        v8f a[2][8];
#pragma unroll
        for (int mb = 0; mb < 2; ++mb)
#pragma unroll
            for (int nb = 0; nb < 8; ++nb) a[mb][nb] = v8f{};

        const _Float16* wbase = W1h + (size_t)(nc8 * 128 + col) * 512 + khi * 16;
#pragma unroll 2
        for (int kc = 0; kc < 16; ++kc) {         // K = 512 in chunks of 32
            // A fragments (LDS), reused across 8 B columns
            const _Float16* g0 = gbuf + col * 512 + kc * 32 + khi * 16;
            v16h A0 = cat8(*(const half8*)g0, *(const half8*)(g0 + 8));
            const _Float16* g1 = g0 + 16 * 512;
            v16h A1 = cat8(*(const half8*)g1, *(const half8*)(g1 + 8));
#pragma unroll
            for (int nb = 0; nb < 8; ++nb) {
                v16h Bf = *(const v16h*)(wbase + (size_t)(nb * 16) * 512 + kc * 32);
                a[0][nb] = __builtin_amdgcn_wmma_f32_16x16x32_f16(false, A0, false, Bf,
                                                                  (short)0, a[0][nb], false, false);
                a[1][nb] = __builtin_amdgcn_wmma_f32_16x16x32_f16(false, A1, false, Bf,
                                                                  (short)0, a[1][nb], false, false);
            }
        }

        // --- prefetch next 128KB W1 slab (overlaps with gelu + GEMM2) ---
        if (nc8 < 3) {
            const char* p = (const char*)(W1h + (size_t)((nc8 + 1) * 128 + (threadIdx.x >> 1)) * 512)
                            + (threadIdx.x & 1) * 512;
            __builtin_prefetch(p, 0, 1);
            __builtin_prefetch(p + 256, 0, 1);
        }

        // --- gelu + h round-trip + GEMM2, in two 64-col halves ---
#pragma unroll
        for (int half64 = 0; half64 < 2; ++half64) {
            // bias + exact gelu -> h[32][64] (LDS, wave-private)
#pragma unroll
            for (int nb4 = 0; nb4 < 4; ++nb4) {
                const int nb = half64 * 4 + nb4;
                float bias = b1[nc8 * 128 + nb * 16 + col];
#pragma unroll
                for (int mb = 0; mb < 2; ++mb) {
#pragma unroll
                    for (int v = 0; v < 8; ++v) {
                        float x0 = gelu_exact(a[mb][nb][v] + bias);
                        hbuf[(mb * 16 + khi * 8 + v) * 64 + nb4 * 16 + col] = (_Float16)x0;
                    }
                }
            }
            // GEMM2: out(32x32) += h(32x64) @ W2^T slab
            const int kg = nc8 * 128 + half64 * 64;
#pragma unroll
            for (int kc2 = 0; kc2 < 2; ++kc2) {
#pragma unroll
                for (int mb = 0; mb < 2; ++mb) {
                    const _Float16* hp = hbuf + (mb * 16 + col) * 64 + kc2 * 32 + khi * 16;
                    v16h Ah = cat8(*(const half8*)hp, *(const half8*)(hp + 8));
#pragma unroll
                    for (int hc = 0; hc < 2; ++hc) {
                        v16h Bf = *(const v16h*)(W2h + (size_t)(hc * 16 + col) * 512
                                                 + kg + kc2 * 32 + khi * 16);
                        acc2[mb * 2 + hc] = __builtin_amdgcn_wmma_f32_16x16x32_f16(
                            false, Ah, false, Bf, (short)0, acc2[mb * 2 + hc], false, false);
                    }
                }
            }
        }
    }

    // ---- Epilogue: bias, LDS transpose stage, coalesced head-major store ----
#pragma unroll
    for (int mb = 0; mb < 2; ++mb) {
#pragma unroll
        for (int hc = 0; hc < 2; ++hc) {
            float bias = b2[hc * 16 + col];
#pragma unroll
            for (int v = 0; v < 8; ++v)
                obuf[(mb * 16 + khi * 8 + v) * 32 + hc * 16 + col] =
                    acc2[mb * 2 + hc][v] + bias;
        }
    }
    // out[b][o][s]: b = tok0/65536, s = (tok0%65536)+row; lanes write 128B runs
    const int bidx = tok0 >> 16;
    const int s0   = tok0 & 65535;
    float* obase = out + (size_t)bidx * (32 * 65536) + s0 + ln;
#pragma unroll 4
    for (int o = 0; o < 32; ++o)
        obase[(size_t)o * 65536] = obuf[ln * 32 + o];
}

// ---------------- launcher ----------------

extern "C" void kernel_launch(void* const* d_in, const int* in_sizes, int n_in,
                              void* d_out, int out_size, void* d_ws, size_t ws_size,
                              hipStream_t stream) {
    const float* rpe   = (const float*)d_in[0];
    const float* means = (const float*)d_in[1];
    const float* stds  = (const float*)d_in[2];
    const float* W1    = (const float*)d_in[3];
    const float* b1    = (const float*)d_in[4];
    const float* W2    = (const float*)d_in[5];
    const float* b2    = (const float*)d_in[6];
    float* out = (float*)d_out;

    _Float16* W1h = (_Float16*)d_ws;                 // 512*512 f16 = 512 KB
    _Float16* W2h = W1h + 512 * 512;                 // 32*512 f16  =  32 KB
    float* gmean  = (float*)(W2h + 32 * 512);
    float* gistd  = gmean + 512;
    float* gscl   = gistd + 512;

    hipLaunchKernelGGL(cvt_f32_f16, dim3(1024), dim3(256), 0, stream, W1, W1h, 512 * 512);
    hipLaunchKernelGGL(cvt_f32_f16, dim3(64),   dim3(256), 0, stream, W2, W2h, 32 * 512);
    hipLaunchKernelGGL(prep_params, dim3(2),    dim3(256), 0, stream,
                       means, stds, gmean, gistd, gscl);

    // T = 4*256*256 = 262144 tokens, 256 per block
    hipLaunchKernelGGL(rpe_fused, dim3(262144 / 256), dim3(256), 0, stream,
                       rpe, W1h, b1, W2h, b2, gmean, gistd, gscl, out);
}
